// SelectivePruningModel_15418932592953
// MI455X (gfx1250) — matrix operand order repaired and checked by reading, not compiled, not run on hardware
//
#include <hip/hip_runtime.h>
#include <hip/hip_bf16.h>

// ---------------------------------------------------------------------------
// Model dims (fixed by the reference)
// ---------------------------------------------------------------------------
#define MROWS 16384   // 8 * 2048 tokens
#define DMODEL 1024
#define DFF    4096
#define DIN    128
#define VOCAB  64
#define NLAYER 4
#define KKEEP  512

typedef __attribute__((ext_vector_type(8)))  __bf16 v8bf;
typedef __attribute__((ext_vector_type(16))) __bf16 v16bf;
typedef __attribute__((ext_vector_type(8)))  float  v8f;

// ---------------------------------------------------------------------------
// fp32 -> bf16 elementwise convert
// ---------------------------------------------------------------------------
__global__ __launch_bounds__(256)
void cvt_f32_bf16(const float* __restrict__ in, __bf16* __restrict__ out, int n) {
    int i = blockIdx.x * 256 + threadIdx.x;
    if (i < n) out[i] = (__bf16)in[i];
}

// ---------------------------------------------------------------------------
// Transpose + convert: W [Kd][Nd] fp32 (row-major) -> Wt [Nd][Kd] bf16.
// 32x32 LDS tile, block (32,8). Kd, Nd multiples of 32.
// ---------------------------------------------------------------------------
__global__ __launch_bounds__(256)
void transpose_cvt(const float* __restrict__ W, __bf16* __restrict__ Wt,
                   int Kd, int Nd) {
    __shared__ float tile[32][33];
    int nb = blockIdx.x * 32;
    int kb = blockIdx.y * 32;
    int tx = threadIdx.x;          // 0..31 (fast, along N for loads)
    int ty = threadIdx.y;          // 0..7
#pragma unroll
    for (int r = 0; r < 4; ++r) {
        int k = kb + ty + r * 8;
        tile[ty + r * 8][tx] = W[(size_t)k * Nd + nb + tx];
    }
    __syncthreads();
#pragma unroll
    for (int r = 0; r < 4; ++r) {
        int n = nb + ty + r * 8;
        Wt[(size_t)n * Kd + kb + tx] = (__bf16)tile[tx][ty + r * 8];
    }
}

// ---------------------------------------------------------------------------
// WMMA bf16 GEMM:  C[M][N] = epilogue(A[M][K] @ Bt[N][K]^T + bias)
//   A  : bf16 row-major [M][K]
//   Bt : bf16 row-major [N][K]  (i.e. B transposed)
//   bias: fp32 [N] (may be null)
//   residual: fp32 [M][N] added pre-activation (may be null; may alias Cf)
//   act_silu: apply x*sigmoid(x)
//   Cb != null -> write bf16, else write fp32 to Cf.
// Block: 128 threads (4 waves). Block tile 256(M) x 64(N); each wave: 64 x 64
// (4 M-tiles x 4 N-tiles -> 16 WMMAs per 8 fragment loads per K-step).
// M multiple of 256, N multiple of 64, K multiple of 32.
// ---------------------------------------------------------------------------
__global__ __launch_bounds__(128)
void gemm_bf16_wmma(const __bf16* __restrict__ A, const __bf16* __restrict__ Bt,
                    const float* __restrict__ bias, const float* __restrict__ residual,
                    float* __restrict__ Cf, __bf16* __restrict__ Cb,
                    int M, int N, int K, int act_silu) {
    const int wave = threadIdx.x >> 5;
    const int lane = threadIdx.x & 31;
    const int lr   = lane & 15;     // row (A) / col (B) within 16
    const int hh   = lane >> 4;     // K-half selector

    const int m0 = blockIdx.y * 256 + wave * 64;   // wave owns rows m0..m0+63
    const int n0 = blockIdx.x * 64;

    v8f acc[4][4];
#pragma unroll
    for (int mt = 0; mt < 4; ++mt)
#pragma unroll
        for (int t = 0; t < 4; ++t) acc[mt][t] = (v8f)(0.0f);

    const __bf16* arow[4];
#pragma unroll
    for (int mt = 0; mt < 4; ++mt)
        arow[mt] = A + (size_t)(m0 + mt * 16 + lr) * K;

    for (int k = 0; k < K; k += 32) {
        // A fragments: lane holds contiguous K[16*hh..16*hh+15] (matches B)
        v16bf afrag[4];
#pragma unroll
        for (int mt = 0; mt < 4; ++mt) {
            v8bf alo = *(const v8bf*)(arow[mt] + k + hh * 16);
            v8bf ahi = *(const v8bf*)(arow[mt] + k + hh * 16 + 8);
            afrag[mt] = __builtin_shufflevector(alo, ahi,
                0, 1, 2, 3, 4, 5, 6, 7, 8, 9, 10, 11, 12, 13, 14, 15);
        }
#pragma unroll
        for (int t = 0; t < 4; ++t) {
            // B fragment: lane holds column n, contiguous K[16*hh..16*hh+15]
            const __bf16* bp = Bt + (size_t)(n0 + t * 16 + lr) * K + k + hh * 16;
            v16bf bfrag = *(const v16bf*)bp;
#pragma unroll
            for (int mt = 0; mt < 4; ++mt) {
                acc[mt][t] = __builtin_amdgcn_wmma_f32_16x16x32_bf16(
                    false, afrag[mt], false, bfrag, (short)0, acc[mt][t], false, false);
            }
        }
    }

    // Epilogue. C/D layout: lane lr = N, vgpr r -> M = r + 8*hh.
#pragma unroll
    for (int mt = 0; mt < 4; ++mt) {
#pragma unroll
        for (int t = 0; t < 4; ++t) {
            int n = n0 + t * 16 + lr;
            float bv = bias ? bias[n] : 0.0f;
#pragma unroll
            for (int r = 0; r < 8; ++r) {
                int m = m0 + mt * 16 + hh * 8 + r;
                size_t idx = (size_t)m * N + n;
                float v = acc[mt][t][r] + bv;
                if (residual) v += residual[idx];
                if (act_silu) v = v / (1.0f + __expf(-v));   // silu
                if (Cb) Cb[idx] = (__bf16)v;
                else    Cf[idx] = v;
            }
        }
    }
}

// ---------------------------------------------------------------------------
// LayerNorm over last dim (1024) -> bf16 output.  One block (256 thr) per row.
// Biased variance, eps = 1e-5 (matches jnp.var / reference _ln).
// ---------------------------------------------------------------------------
__global__ __launch_bounds__(256)
void ln_to_bf16(const float* __restrict__ x, const float* __restrict__ g,
                const float* __restrict__ b, __bf16* __restrict__ out) {
    __shared__ float rs[256];
    __shared__ float rq[256];
    const int row = blockIdx.x;
    const int tid = threadIdx.x;
    const float* xr = x + (size_t)row * DMODEL;

    float v[4];
    float s = 0.0f, q = 0.0f;
#pragma unroll
    for (int t = 0; t < 4; ++t) {
        v[t] = xr[tid + t * 256];
        s += v[t];
        q += v[t] * v[t];
    }
    rs[tid] = s; rq[tid] = q;
    __syncthreads();
    for (int off = 128; off > 0; off >>= 1) {
        if (tid < off) { rs[tid] += rs[tid + off]; rq[tid] += rq[tid + off]; }
        __syncthreads();
    }
    const float inv = 1.0f / (float)DMODEL;
    float mean = rs[0] * inv;
    float var  = rq[0] * inv - mean * mean;
    float rstd = rsqrtf(var + 1e-5f);
#pragma unroll
    for (int t = 0; t < 4; ++t) {
        int i = tid + t * 256;
        out[(size_t)row * DMODEL + i] = (__bf16)((v[t] - mean) * rstd * g[i] + b[i]);
    }
}

// ---------------------------------------------------------------------------
// Selective pruning: per row, keep x[i] where |x[i]| >= (KKEEP-th largest |x|).
// Bitonic sort of |x| (1024 floats) in LDS; threshold = sorted_asc[1024-KKEEP].
// ---------------------------------------------------------------------------
__global__ __launch_bounds__(256)
void prune_topk(float* __restrict__ x) {
    __shared__ float s[DMODEL];
    const int row = blockIdx.x;
    const int tid = threadIdx.x;
    float* xr = x + (size_t)row * DMODEL;

#pragma unroll
    for (int t = 0; t < 4; ++t) {
        int i = tid + t * 256;
        s[i] = fabsf(xr[i]);
    }
    __syncthreads();

    for (int k2 = 2; k2 <= DMODEL; k2 <<= 1) {
        for (int j = k2 >> 1; j > 0; j >>= 1) {
#pragma unroll
            for (int t = 0; t < 4; ++t) {
                int i = tid + t * 256;
                int l = i ^ j;
                if (l > i) {
                    bool up = ((i & k2) == 0);
                    float a = s[i], c = s[l];
                    if ((a > c) == up) { s[i] = c; s[l] = a; }
                }
            }
            __syncthreads();
        }
    }
    float thr = s[DMODEL - KKEEP];   // KKEEP-th largest
#pragma unroll
    for (int t = 0; t < 4; ++t) {
        int i = tid + t * 256;
        float v = xr[i];
        xr[i] = (fabsf(v) >= thr) ? v : 0.0f;
    }
}

// ---------------------------------------------------------------------------
// Host-side orchestration
// ---------------------------------------------------------------------------
extern "C" void kernel_launch(void* const* d_in, const int* in_sizes, int n_in,
                              void* d_out, int out_size, void* d_ws, size_t ws_size,
                              hipStream_t stream) {
    const float* frames  = (const float*)d_in[0];
    const float* W_in    = (const float*)d_in[1];
    const float* b_in    = (const float*)d_in[2];
    const float* ln_g    = (const float*)d_in[3];
    const float* ln_b    = (const float*)d_in[4];
    const float* W1      = (const float*)d_in[5];
    const float* b1      = (const float*)d_in[6];
    const float* W2      = (const float*)d_in[7];
    const float* b2      = (const float*)d_in[8];
    const float* fn_g    = (const float*)d_in[9];
    const float* fn_b    = (const float*)d_in[10];
    const float* W_frame = (const float*)d_in[11];
    const float* b_frame = (const float*)d_in[12];
    const float* W_sym   = (const float*)d_in[13];
    const float* b_sym   = (const float*)d_in[14];

    // Workspace carve-up (256B aligned segments)
    char* wsp = (char*)d_ws;
    auto alloc = [&](size_t bytes) -> char* {
        char* p = wsp;
        wsp += (bytes + 255) & ~(size_t)255;
        return p;
    };
    float*  x    = (float*) alloc((size_t)MROWS * DMODEL * 4);
    __bf16* xn   = (__bf16*)alloc((size_t)MROWS * DMODEL * 2);
    __bf16* h    = (__bf16*)alloc((size_t)MROWS * DFF    * 2);
    __bf16* fb   = (__bf16*)alloc((size_t)MROWS * DIN    * 2);
    __bf16* WtIn = (__bf16*)alloc((size_t)DMODEL * DIN   * 2);
    __bf16* W1t  = (__bf16*)alloc((size_t)NLAYER * DFF * DMODEL * 2);
    __bf16* W2t  = (__bf16*)alloc((size_t)NLAYER * DMODEL * DFF * 2);
    __bf16* WfT  = (__bf16*)alloc((size_t)DIN   * DMODEL * 2);
    __bf16* WsT  = (__bf16*)alloc((size_t)VOCAB * DMODEL * 2);

    dim3 tb(32, 8);
    // Weight conversions: W [K][N] -> Wt [N][K] bf16
    transpose_cvt<<<dim3(DMODEL / 32, DIN / 32), tb, 0, stream>>>(W_in, WtIn, DIN, DMODEL);
    for (int l = 0; l < NLAYER; ++l) {
        transpose_cvt<<<dim3(DFF / 32, DMODEL / 32), tb, 0, stream>>>(
            W1 + (size_t)l * DMODEL * DFF, W1t + (size_t)l * DFF * DMODEL, DMODEL, DFF);
        transpose_cvt<<<dim3(DMODEL / 32, DFF / 32), tb, 0, stream>>>(
            W2 + (size_t)l * DFF * DMODEL, W2t + (size_t)l * DMODEL * DFF, DFF, DMODEL);
    }
    transpose_cvt<<<dim3(DIN / 32, DMODEL / 32), tb, 0, stream>>>(W_frame, WfT, DMODEL, DIN);
    transpose_cvt<<<dim3(VOCAB / 32, DMODEL / 32), tb, 0, stream>>>(W_sym, WsT, DMODEL, VOCAB);

    cvt_f32_bf16<<<(MROWS * DIN) / 256, 256, 0, stream>>>(frames, fb, MROWS * DIN);

    // Input projection: x = frames @ W_in + b_in   (fp32 out)
    gemm_bf16_wmma<<<dim3(DMODEL / 64, MROWS / 256), 128, 0, stream>>>(
        fb, WtIn, b_in, nullptr, x, nullptr, MROWS, DMODEL, DIN, 0);

    for (int l = 0; l < NLAYER; ++l) {
        // xn = LN(x) (bf16)
        ln_to_bf16<<<MROWS, 256, 0, stream>>>(x, ln_g + (size_t)l * DMODEL,
                                              ln_b + (size_t)l * DMODEL, xn);
        // h = silu(xn @ W1 + b1)  (bf16 out)
        gemm_bf16_wmma<<<dim3(DFF / 64, MROWS / 256), 128, 0, stream>>>(
            xn, W1t + (size_t)l * DFF * DMODEL, b1 + (size_t)l * DFF,
            nullptr, nullptr, h, MROWS, DFF, DMODEL, 1);
        // x = x + h @ W2 + b2  (fp32 out, in-place residual)
        gemm_bf16_wmma<<<dim3(DMODEL / 64, MROWS / 256), 128, 0, stream>>>(
            h, W2t + (size_t)l * DMODEL * DFF, b2 + (size_t)l * DMODEL,
            x, x, nullptr, MROWS, DMODEL, DFF, 0);
        // selective pruning (top-512 by |x| per row)
        prune_topk<<<MROWS, 256, 0, stream>>>(x);
    }

    // Final norm + heads
    ln_to_bf16<<<MROWS, 256, 0, stream>>>(x, fn_g, fn_b, xn);
    float* out_frame = (float*)d_out;
    float* out_sym   = (float*)d_out + (size_t)MROWS * DIN;
    gemm_bf16_wmma<<<dim3(DIN / 64, MROWS / 256), 128, 0, stream>>>(
        xn, WfT, b_frame, nullptr, out_frame, nullptr, MROWS, DIN, DMODEL, 0);
    gemm_bf16_wmma<<<dim3(VOCAB / 64, MROWS / 256), 128, 0, stream>>>(
        xn, WsT, b_sym, nullptr, out_sym, nullptr, MROWS, VOCAB, DMODEL, 0);
}
